// SSM_69080253989410
// MI455X (gfx1250) — matrix operand order repaired and checked by reading, not compiled, not run on hardware
//
#include <hip/hip_runtime.h>
#include <hip/hip_bf16.h>
#include <cstdint>
#include <cstddef>

// ---------------------------------------------------------------------------
// Problem constants (from reference)
// ---------------------------------------------------------------------------
#define D_MODEL   1024
#define D_STATE   128
#define D_INNER   2048
#define NHEADS    32
#define HEADDIM   64
#define CONV_DIM  2176
#define D_IN_PROJ 4256              // 2*2048 + 128 + 32
#define NB        4
#define LSEQ      4096
#define ROWS      (NB * LSEQ)       // 16384
#define DT_OFF    (D_INNER + CONV_DIM)  // 4224
#define CLEN      128               // scan chunk length
#define NCH       (LSEQ / CLEN)     // 32 chunks per sequence

typedef __bf16 bf16_t;
typedef __attribute__((ext_vector_type(16))) __bf16        v16bf;
typedef __attribute__((ext_vector_type(8)))  float         v8f;
typedef __attribute__((ext_vector_type(8)))  unsigned int  v8u;
typedef int v4i_t __attribute__((vector_size(16)));        // matches builtin

#define AS1 __attribute__((address_space(1)))
#define AS3 __attribute__((address_space(3)))

// ---------------------------------------------------------------------------
// CDNA5 async global->LDS copy (GLOBAL_LOAD_ASYNC_TO_LDS_B128, ASYNCcnt)
// ---------------------------------------------------------------------------
static __device__ __forceinline__ void async_copy_b128(const void* gp, void* lp) {
#if __has_builtin(__builtin_amdgcn_global_load_async_to_lds_b128)
  __builtin_amdgcn_global_load_async_to_lds_b128(
      (AS1 v4i_t*)gp, (AS3 v4i_t*)lp, 0, 0);
#else
  unsigned lo = (unsigned)(__SIZE_TYPE__)(AS3 void*)lp;
  asm volatile("global_load_async_to_lds_b128 %0, %1, off"
               :: "v"(lo), "v"(gp) : "memory");
#endif
}

static __device__ __forceinline__ void wait_async0() {
#if __has_builtin(__builtin_amdgcn_s_wait_asynccnt)
  __builtin_amdgcn_s_wait_asynccnt(0);
#else
  asm volatile("s_wait_asynccnt 0x0" ::: "memory");
#endif
}

// ---------------------------------------------------------------------------
// f32 -> bf16 conversion (RNE via compiler cast)
// ---------------------------------------------------------------------------
__global__ void f32_to_bf16_kernel(const float* __restrict__ in,
                                   bf16_t* __restrict__ out, long n) {
  long i = (long)blockIdx.x * blockDim.x + threadIdx.x;
  if (i < n) out[i] = (bf16_t)in[i];
}

// ---------------------------------------------------------------------------
// Tiny reduction: S0 = sum(bb), S1 = sum(bw[:,0])  (C==ones collapse)
// ---------------------------------------------------------------------------
__global__ void scalars_kernel(const float* __restrict__ bw,
                               const float* __restrict__ bb,
                               float* __restrict__ scal) {
  if (threadIdx.x == 0 && blockIdx.x == 0) {
    float s0 = 0.f, s1 = 0.f;
    for (int i = 0; i < D_STATE; ++i) { s0 += bb[i]; s1 += bw[i]; }
    scal[0] = s0; scal[1] = s1;
  }
}

// ---------------------------------------------------------------------------
// bf16 TN GEMM with f32 accumulate:
//   C[m,n] = sum_k A[m*lda+k] * B[n*ldb+k] (+ bias[n])
// Block tile 128x128, BK=32, 256 threads = 8 waves (4x2), wave tile 32x64
// (2x4 WMMA fragments).  Staging via GLOBAL_LOAD_ASYNC_TO_LDS_B128, double
// buffered; V_WMMA_F32_16X16X32_BF16 compute.  LDS rows padded to 40 halves
// (80B) -> conflict-free fragment gathers (lid*20 mod 64 distinct) while
// keeping b128 alignment.
// ---------------------------------------------------------------------------
#define BM 128
#define BN 128
#define BK 32
#define LDSK 40

__global__ __launch_bounds__(256)
void gemm_bf16_tn(const bf16_t* __restrict__ A, const bf16_t* __restrict__ Bmat,
                  float* __restrict__ C, const float* __restrict__ bias,
                  int N, int K, int lda, int ldb, int ldc)
{
  __shared__ bf16_t As[2][BM][LDSK];
  __shared__ bf16_t Bs[2][BN][LDSK];

  const int tid  = threadIdx.x;
  const int wave = tid >> 5;
  const int lane = tid & 31;
  const int lid  = lane & 15;   // row/col within 16
  const int hi   = lane >> 4;   // lane-half selects K range
  const int wm   = wave >> 1;   // 0..3 -> rows wm*32
  const int wn   = wave & 1;    // 0..1 -> cols wn*64
  const long m0  = (long)blockIdx.y * BM;
  const int  n0  = blockIdx.x * BN;

  // staging coordinates: each thread DMAs 4 x 16B per stage
  const int grow = tid >> 2;          // 0..63
  const int gcol = (tid & 3) << 3;    // 0,8,16,24

  // clamp OOB B rows: they only feed accumulators whose columns are never
  // stored (epilogue guards col < N)
  const int bn0 = min(n0 + grow, N - 1);
  const int bn1 = min(n0 + grow + 64, N - 1);

  const bf16_t* a0 = A + (m0 + grow) * (long)lda + gcol;
  const bf16_t* a1 = a0 + 64l * lda;
  const bf16_t* b0 = Bmat + (long)bn0 * ldb + gcol;
  const bf16_t* b1 = Bmat + (long)bn1 * ldb + gcol;

  v8f acc[2][4];
#pragma unroll
  for (int i = 0; i < 2; ++i)
#pragma unroll
    for (int j = 0; j < 4; ++j)
#pragma unroll
      for (int e = 0; e < 8; ++e) acc[i][j][e] = 0.f;

  // ---- prologue: stage 0 ----
  async_copy_b128(a0, &As[0][grow][gcol]);
  async_copy_b128(a1, &As[0][grow + 64][gcol]);
  async_copy_b128(b0, &Bs[0][grow][gcol]);
  async_copy_b128(b1, &Bs[0][grow + 64][gcol]);
  wait_async0();
  __syncthreads();

  const int nk = K / BK;
  for (int kk = 0; kk < nk; ++kk) {
    const int  cur  = kk & 1;
    const bool more = (kk + 1 < nk);

    if (more) {  // async DMA next stage into the other buffer
      const long ko = (long)(kk + 1) * BK;
      async_copy_b128(a0 + ko, &As[cur ^ 1][grow][gcol]);
      async_copy_b128(a1 + ko, &As[cur ^ 1][grow + 64][gcol]);
      async_copy_b128(b0 + ko, &Bs[cur ^ 1][grow][gcol]);
      async_copy_b128(b1 + ko, &Bs[cur ^ 1][grow + 64][gcol]);
    }

    // ---- fragment gathers per CDNA5 16-bit WMMA lane layout ----
    const unsigned* Au = (const unsigned*)&As[cur][0][0];
    const unsigned* Bu = (const unsigned*)&Bs[cur][0][0];
    v8u afr[2], bfr[4];
#pragma unroll
    for (int mt = 0; mt < 2; ++mt) {
      const int row = wm * 32 + mt * 16 + lid;
#pragma unroll
      for (int v = 0; v < 8; ++v) {
        // A 16x32: lanes0-15 K=0..7 (v0..3),16..23 (v4..7); lanes16-31 +8
        const int k0 = (v < 4) ? (8 * hi + 2 * v) : (16 + 8 * hi + 2 * (v - 4));
        afr[mt][v] = Au[row * (LDSK / 2) + (k0 >> 1)];
      }
    }
#pragma unroll
    for (int nt = 0; nt < 4; ++nt) {
      const int col = wn * 64 + nt * 16 + lid;
#pragma unroll
      for (int v = 0; v < 8; ++v) {
        // B 32x16: lanes0-15 K=0..7 (v0..3),16..23 (v4..7); lanes16-31 +8
        const int k0 = (v < 4) ? (8 * hi + 2 * v) : (16 + 8 * hi + 2 * (v - 4));
        bfr[nt][v] = Bu[col * (LDSK / 2) + (k0 >> 1)];
      }
    }

#pragma unroll
    for (int mt = 0; mt < 2; ++mt)
#pragma unroll
      for (int nt = 0; nt < 4; ++nt)
        acc[mt][nt] = __builtin_amdgcn_wmma_f32_16x16x32_bf16(
            false, __builtin_bit_cast(v16bf, afr[mt]),
            false, __builtin_bit_cast(v16bf, bfr[nt]),
            (short)0, acc[mt][nt], false, false);

    if (more) wait_async0();  // own asyncs done before barrier
    __syncthreads();          // all waves' stage kk+1 now visible
  }

  // ---- epilogue: D layout = VGPR i -> row (i + 8*hi), col lid ----
#pragma unroll
  for (int mt = 0; mt < 2; ++mt)
#pragma unroll
    for (int nt = 0; nt < 4; ++nt) {
      const int col = n0 + wn * 64 + nt * 16 + lid;
      if (col < N) {
        const float bv = bias ? bias[col] : 0.f;
#pragma unroll
        for (int i = 0; i < 8; ++i) {
          const long row = m0 + wm * 32 + mt * 16 + hi * 8 + i;
          C[row * (long)ldc + col] = acc[mt][nt][i] + bv;
        }
      }
    }
}

// ---------------------------------------------------------------------------
// dt = softplus(raw + dt_bias), a = -exp(A_log)*dt, Bsum = reward*S1 + S0
// ---------------------------------------------------------------------------
__global__ void dt_kernel(const float* __restrict__ zx,
                          const float* __restrict__ reward,
                          const float* __restrict__ dt_bias,
                          const float* __restrict__ A_log,
                          const float* __restrict__ scal,
                          float* __restrict__ dt_arr,
                          float* __restrict__ a_arr,
                          float* __restrict__ bsum) {
  const int tid = threadIdx.x;
  const int row = blockIdx.x * 8 + (tid >> 5);
  const int h   = tid & 31;
  const float raw = zx[(long)row * D_IN_PROJ + DT_OFF + h] + dt_bias[h];
  const float dtv = (raw > 20.f) ? raw : log1pf(__expf(raw));
  dt_arr[row * NHEADS + h] = dtv;
  a_arr[row * NHEADS + h]  = -__expf(A_log[h]) * dtv;
  if (h == 0) bsum[row] = reward[row] * scal[1] + scal[0];
}

// ---------------------------------------------------------------------------
// Causal depthwise conv (taps 4) + custom silu (sigmoid(x)+0.1x) + *dt
// Only the first D_INNER conv channels are ever consumed downstream.
// ---------------------------------------------------------------------------
__global__ void conv_kernel(const float* __restrict__ zx,
                            const float* __restrict__ conv_w,
                            const float* __restrict__ conv_b,
                            const float* __restrict__ dt_arr,
                            float* __restrict__ xd) {
  const long idx = (long)blockIdx.x * 256 + threadIdx.x;  // < ROWS*D_INNER
  const int  c   = (int)(idx & (D_INNER - 1));
  const long row = idx >> 11;
  const int  l   = (int)(row & (LSEQ - 1));
  float acc = conv_b[c];
#pragma unroll
  for (int w = 0; w < 4; ++w) {
    const int ls = l - 3 + w;
    const float v = (ls >= 0)
        ? zx[(row + (w - 3)) * (long)D_IN_PROJ + D_INNER + c] : 0.f;
    acc += v * conv_w[c * 4 + w];
  }
  const float x = 1.f / (1.f + __expf(-acc)) + 0.1f * acc;
  xd[idx] = x * dt_arr[row * NHEADS + (c >> 6)];
}

// ---------------------------------------------------------------------------
// Scan pass 1: per-chunk local recurrence (in place over xd), plus chunk
// summaries.  C==ones collapses D_STATE -> scalar state per (b,h,p).
// ---------------------------------------------------------------------------
__global__ __launch_bounds__(256)
void scan1_kernel(const float* __restrict__ a_arr,
                  const float* __restrict__ bsum,
                  float* __restrict__ xd,        // in: x*dt, out: local Y
                  float* __restrict__ acum,      // cumsum(a) within chunk
                  float* __restrict__ Ech,       // exp(chunk-total a)
                  float* __restrict__ Send) {    // chunk-final local state
  const int b    = blockIdx.z;                   // 4
  const int cidx = blockIdx.y;                   // NCH
  const int h    = blockIdx.x * 4 + (threadIdx.x >> 6);  // 4 heads/block
  const int p    = threadIdx.x & 63;
  const int l0   = cidx * CLEN;
  float S = 0.f, asum = 0.f;
  for (int t = 0; t < CLEN; ++t) {
    const long row = (long)b * LSEQ + l0 + t;
    const float av = a_arr[row * NHEADS + h];
    const float e  = __expf(av);
    const float bs = bsum[row];
    const long  xi = row * D_INNER + h * HEADDIM + p;
    S = e * S + bs * xd[xi];
    xd[xi] = S;
    asum += av;
    if (p == 0) acum[row * NHEADS + h] = asum;
  }
  const long co = (long)(b * NCH + cidx) * NHEADS + h;
  if (p == 0) Ech[co] = __expf(asum);
  Send[co * HEADDIM + p] = S;
}

// ---------------------------------------------------------------------------
// Scan pass 2: sequential carry across NCH chunks per (b,h,p).
// ---------------------------------------------------------------------------
__global__ void scan2_kernel(const float* __restrict__ Ech,
                             const float* __restrict__ Send,
                             float* __restrict__ Ppre) {
  const int idx = blockIdx.x * 256 + threadIdx.x;  // < NB*NHEADS*HEADDIM
  const int p = idx & 63;
  const int h = (idx >> 6) & 31;
  const int b = idx >> 11;
  float carry = 0.f;
  for (int c = 0; c < NCH; ++c) {
    const long o = (long)(b * NCH + c) * NHEADS + h;
    Ppre[o * HEADDIM + p] = carry;
    carry = Ech[o] * carry + Send[o * HEADDIM + p];
  }
}

// ---------------------------------------------------------------------------
// Pass 3 fused: Y = Yloc + carry_in*exp(acum); gate with silu(z); pack bf16.
// ---------------------------------------------------------------------------
__global__ void finalize_kernel(const float* __restrict__ xd,    // local Y
                                const float* __restrict__ Ppre,
                                const float* __restrict__ acum,
                                const float* __restrict__ zx,
                                bf16_t* __restrict__ ybf) {
  const long idx = (long)blockIdx.x * 256 + threadIdx.x;  // < ROWS*D_INNER
  const int  c   = (int)(idx & (D_INNER - 1));
  const long row = idx >> 11;
  const int  l   = (int)(row & (LSEQ - 1));
  const int  b   = (int)(row >> 12);
  const int  h   = c >> 6, p = c & 63;
  const int  cidx = l / CLEN;
  const long co = (long)(b * NCH + cidx) * NHEADS + h;
  const float Y = xd[idx] + Ppre[co * HEADDIM + p] * __expf(acum[row * NHEADS + h]);
  const float z = zx[row * (long)D_IN_PROJ + c];
  const float g = 1.f / (1.f + __expf(-z)) + 0.1f * z;
  ybf[idx] = (bf16_t)(Y * g);
}

// ---------------------------------------------------------------------------
// Host launcher
// ---------------------------------------------------------------------------
extern "C" void kernel_launch(void* const* d_in, const int* in_sizes, int n_in,
                              void* d_out, int out_size, void* d_ws, size_t ws_size,
                              hipStream_t stream) {
  const float* u       = (const float*)d_in[0];
  const float* reward  = (const float*)d_in[1];
  const float* in_w    = (const float*)d_in[2];
  const float* in_b    = (const float*)d_in[3];
  const float* conv_w  = (const float*)d_in[4];
  const float* conv_b  = (const float*)d_in[5];
  const float* bw      = (const float*)d_in[6];
  const float* bb      = (const float*)d_in[7];
  const float* dt_bias = (const float*)d_in[8];
  const float* A_log   = (const float*)d_in[9];
  const float* out_w   = (const float*)d_in[10];
  float* out = (float*)d_out;

  // bump allocator on workspace
  size_t off = 0;
  auto wsa = [&](size_t bytes) -> void* {
    void* p = (char*)d_ws + off;
    off = (off + bytes + 255) & ~(size_t)255;
    return p;
  };
  bf16_t* u_bf    = (bf16_t*)wsa((size_t)ROWS * D_MODEL * 2);
  bf16_t* inw_bf  = (bf16_t*)wsa((size_t)D_IN_PROJ * D_MODEL * 2);
  bf16_t* outw_bf = (bf16_t*)wsa((size_t)D_MODEL * D_INNER * 2);
  float*  zx      = (float*) wsa((size_t)ROWS * D_IN_PROJ * 4);
  float*  xd      = (float*) wsa((size_t)ROWS * D_INNER * 4);
  float*  dt_arr  = (float*) wsa((size_t)ROWS * NHEADS * 4);
  float*  a_arr   = (float*) wsa((size_t)ROWS * NHEADS * 4);
  float*  acum    = (float*) wsa((size_t)ROWS * NHEADS * 4);
  float*  bsum    = (float*) wsa((size_t)ROWS * 4);
  float*  scal    = (float*) wsa(2 * 4);
  float*  Ech     = (float*) wsa((size_t)NB * NCH * NHEADS * 4);
  float*  Send    = (float*) wsa((size_t)NB * NCH * NHEADS * HEADDIM * 4);
  float*  Ppre    = (float*) wsa((size_t)NB * NCH * NHEADS * HEADDIM * 4);
  bf16_t* ybf     = (bf16_t*)wsa((size_t)ROWS * D_INNER * 2);
  (void)ws_size; (void)n_in; (void)in_sizes; (void)out_size;

  // bf16 conversions (bandwidth-trivial)
  {
    long n = (long)ROWS * D_MODEL;
    f32_to_bf16_kernel<<<(n + 255) / 256, 256, 0, stream>>>(u, u_bf, n);
    n = (long)D_IN_PROJ * D_MODEL;
    f32_to_bf16_kernel<<<(n + 255) / 256, 256, 0, stream>>>(in_w, inw_bf, n);
    n = (long)D_MODEL * D_INNER;
    f32_to_bf16_kernel<<<(n + 255) / 256, 256, 0, stream>>>(out_w, outw_bf, n);
  }
  scalars_kernel<<<1, 64, 0, stream>>>(bw, bb, scal);

  // GEMM1: zx = u @ in_w.T + in_b   (16384 x 4256, K=1024)
  {
    dim3 grid((D_IN_PROJ + BN - 1) / BN, ROWS / BM);
    gemm_bf16_tn<<<grid, 256, 0, stream>>>(u_bf, inw_bf, zx, in_b,
                                           D_IN_PROJ, D_MODEL,
                                           D_MODEL, D_MODEL, D_IN_PROJ);
  }

  dt_kernel<<<ROWS / 8, 256, 0, stream>>>(zx, reward, dt_bias, A_log, scal,
                                          dt_arr, a_arr, bsum);

  conv_kernel<<<(unsigned)((long)ROWS * D_INNER / 256), 256, 0, stream>>>(
      zx, conv_w, conv_b, dt_arr, xd);

  {
    dim3 grid(NHEADS / 4, NCH, NB);
    scan1_kernel<<<grid, 256, 0, stream>>>(a_arr, bsum, xd, acum, Ech, Send);
  }
  scan2_kernel<<<(NB * NHEADS * HEADDIM) / 256, 256, 0, stream>>>(Ech, Send, Ppre);

  finalize_kernel<<<(unsigned)((long)ROWS * D_INNER / 256), 256, 0, stream>>>(
      xd, Ppre, acum, zx, ybf);

  // GEMM2: out = y_gated @ out_w.T  (16384 x 1024, K=2048)
  {
    dim3 grid(D_MODEL / BN, ROWS / BM);
    gemm_bf16_tn<<<grid, 256, 0, stream>>>(ybf, outw_bf, out, nullptr,
                                           D_MODEL, D_INNER,
                                           D_INNER, D_INNER, D_MODEL);
  }
}
